// GQAttention_90477781057977
// MI455X (gfx1250) — matrix operand order repaired and checked by reading, not compiled, not run on hardware
//
#include <hip/hip_runtime.h>

// ---------------------------------------------------------------------------
// GQA attention block for MI455X (gfx1250), wave32 + WMMA bf16 + TDM.
// B=2, T=2048, D_MODEL=1024, H=16, KV=4, D=64, GROUPS=4.
// ---------------------------------------------------------------------------

#define DEVFN __device__ __forceinline__

typedef __attribute__((ext_vector_type(16))) __bf16 v16bf;
typedef __attribute__((ext_vector_type(8)))  float  v8f;

union BFrag {
  v16bf v;
  uint4 q[2];
};

DEVFN v8f wmma_bf16(v16bf a, v16bf b, v8f c) {
  // D(f32 16x16) = A(bf16 16x32) * B(bf16 32x16) + C
  return __builtin_amdgcn_wmma_f32_16x16x32_bf16(
      /*neg_a=*/false, a, /*neg_b=*/false, b,
      /*c_mod=*/(short)0, c, /*reuse_a=*/false, /*reuse_b=*/false);
}

// ---------------------------------------------------------------------------
// Tensor Data Mover: 2-D bf16 tile load Global -> LDS with LDS row padding.
// ---------------------------------------------------------------------------
#if defined(__has_builtin)
#if __has_builtin(__builtin_amdgcn_tensor_load_to_lds)
#define USE_TDM 1
#endif
#endif
#ifndef USE_TDM
#define USE_TDM 0
#endif

#if USE_TDM
typedef __attribute__((ext_vector_type(4))) unsigned int tdm_v4u;
typedef __attribute__((ext_vector_type(8))) int          tdm_v8i;
typedef __attribute__((ext_vector_type(4))) int          tdm_v4i;

// Loads a (tile_d1 x tile_d0) tile of 2-byte elements starting at gaddr,
// row stride = stride_d0 elements, into LDS at lds_off. Adds pad_amount
// DWORDs of LDS padding every pad_interval DWORDs (codes per D# spec) so the
// LDS layout matches the padded __shared__ arrays used by the consumers.
DEVFN void tdm_load_tile_2d_bf16(unsigned lds_off, unsigned long long gaddr,
                                 unsigned tensor_d0, unsigned tensor_d1,
                                 unsigned tile_d0, unsigned tile_d1,
                                 unsigned stride_d0, unsigned pad_interval_code,
                                 unsigned pad_amount_code, unsigned pad_enable) {
  tdm_v4u g0;
  g0[0] = 1u;                                     // count=1 (valid), user mode
  g0[1] = lds_off;                                // lds_addr (bytes)
  g0[2] = (unsigned)(gaddr & 0xffffffffu);        // global_addr[31:0]
  g0[3] = (unsigned)((gaddr >> 32) & 0x01ffffffu) // global_addr[56:32]
          | (2u << 30);                           // type=2 ("image")
  tdm_v8i g1;
  g1[0] = (int)((1u << 16)                        // data_size=1 -> 2 bytes
                | (pad_enable << 20)              // pad_enable
                | (pad_interval_code << 22)       // pad_interval code
                | (pad_amount_code << 25));       // pad_amount code
  g1[1] = (int)((tensor_d0 & 0xffffu) << 16);     // tensor_dim0[15:0] @ 63:48
  g1[2] = (int)((tensor_d0 >> 16) | ((tensor_d1 & 0xffffu) << 16));
  g1[3] = (int)((tensor_d1 >> 16) | ((tile_d0 & 0xffffu) << 16)); // tile_dim0
  g1[4] = (int)(tile_d1 & 0xffffu);               // tile_dim1 @ 143:128
  g1[5] = (int)stride_d0;                         // tensor_dim0_stride[31:0]
  g1[6] = 0;
  g1[7] = 0;
  tdm_v4i z4 = {0, 0, 0, 0};
#if __clang_major__ >= 23
  tdm_v8i z8 = {0, 0, 0, 0, 0, 0, 0, 0};
  __builtin_amdgcn_tensor_load_to_lds(g0, g1, z4, z4, z8, 0);
#else
  __builtin_amdgcn_tensor_load_to_lds(g0, g1, z4, z4, 0);
#endif
}
#endif  // USE_TDM

// ---------------------------------------------------------------------------
// fp32 -> bf16 elementwise convert
// ---------------------------------------------------------------------------
__global__ __launch_bounds__(256) void cvt_f32_bf16(const float* __restrict__ in,
                                                    __bf16* __restrict__ out, int n) {
  int i = blockIdx.x * 256 + threadIdx.x;
  if (i < n) out[i] = (__bf16)in[i];
}

// ---------------------------------------------------------------------------
// C[M,N](f32) = A[M,K](bf16) @ W[N,K](bf16)^T  -- torch Linear convention.
// Block: 256 threads = 8 waves (4 along M x 2 along N); tile 128x128x64.
// Each wave: 32x64 output = 2x4 WMMA accumulators; 16 WMMA per K-tile.
// ---------------------------------------------------------------------------
__global__ __launch_bounds__(256) void gemm_bf16_wmma(const __bf16* __restrict__ A,
                                                      const __bf16* __restrict__ W,
                                                      float* __restrict__ C,
                                                      int M, int N, int K) {
  constexpr int BM = 128, BN = 128, BK = 64;
  __shared__ __bf16 sA[BM][BK + 8];
  __shared__ __bf16 sW[BN][BK + 8];

  const int tid  = threadIdx.x;
  const int lane = tid & 31;
  const int wave = tid >> 5;
  const int wm   = wave & 3;   // 0..3 along M (32 rows each)
  const int wn   = wave >> 2;  // 0..1 along N (64 cols each)
  const int hgrp = lane >> 4;  // 16-lane half group
  const int l16  = lane & 15;

  const int m0 = blockIdx.y * BM;
  const int n0 = blockIdx.x * BN;

  v8f acc[2][4];
#pragma unroll
  for (int i = 0; i < 2; ++i)
#pragma unroll
    for (int j = 0; j < 4; ++j) acc[i][j] = (v8f){};

  const int srow = tid >> 1;          // 0..127
  const int scol = (tid & 1) * 32;    // 0 / 32

  for (int k0 = 0; k0 < K; k0 += BK) {
    __syncthreads();
    // Stage A tile (128x64) and W tile (128x64); 64B per thread per tile.
    const __bf16* ga = A + (size_t)(m0 + srow) * K + k0 + scol;
    const __bf16* gw = W + (size_t)(n0 + srow) * K + k0 + scol;
#pragma unroll
    for (int e = 0; e < 4; ++e) {
      *(uint4*)&sA[srow][scol + e * 8] = *(const uint4*)(ga + e * 8);
      *(uint4*)&sW[srow][scol + e * 8] = *(const uint4*)(gw + e * 8);
    }
    __builtin_prefetch(ga + BK, 0, 1);   // global_prefetch_b8 for next K tile
    __builtin_prefetch(gw + BK, 0, 1);
    __syncthreads();

#pragma unroll
    for (int ks = 0; ks < BK; ks += 32) {
      // Fragments per the CDNA5 16-bit VGPR layouts.
      BFrag af[2], bf[4];
#pragma unroll
      for (int i = 0; i < 2; ++i) {
        const int r = wm * 32 + i * 16 + l16;                  // A row (M)
        af[i].q[0] = *(const uint4*)&sA[r][ks + hgrp * 8];
        af[i].q[1] = *(const uint4*)&sA[r][ks + 16 + hgrp * 8];
      }
#pragma unroll
      for (int j = 0; j < 4; ++j) {
        const int r = wn * 64 + j * 16 + l16;                  // W row (N)
        bf[j].q[0] = *(const uint4*)&sW[r][ks + hgrp * 8];
        bf[j].q[1] = *(const uint4*)&sW[r][ks + 16 + hgrp * 8];
      }
#pragma unroll
      for (int i = 0; i < 2; ++i)
#pragma unroll
        for (int j = 0; j < 4; ++j)
          acc[i][j] = wmma_bf16(af[i].v, bf[j].v, acc[i][j]);
    }
  }

  // Epilogue: C layout lane -> col = l16, VGPR v -> row = v + 8*hgrp.
#pragma unroll
  for (int i = 0; i < 2; ++i)
#pragma unroll
    for (int j = 0; j < 4; ++j) {
      const int col = n0 + wn * 64 + j * 16 + l16;
#pragma unroll
      for (int v = 0; v < 8; ++v) {
        const int row = m0 + wm * 32 + i * 16 + v + 8 * hgrp;
        C[(size_t)row * N + col] = acc[i][j][v];
      }
    }
}

// ---------------------------------------------------------------------------
// RMSNorm(head_dim) + RoPE + repack to (B, H, T, 64) bf16.
// One block per (b,t); one wave per head; lane owns dims (lane, lane+32).
// ---------------------------------------------------------------------------
__global__ __launch_bounds__(256) void norm_rope_pack(
    const float* __restrict__ qf, const float* __restrict__ kf,
    const float* __restrict__ vf, const float* __restrict__ cosw,
    const float* __restrict__ sinw, const float* __restrict__ qn,
    const float* __restrict__ kn, __bf16* __restrict__ qb,
    __bf16* __restrict__ kb, __bf16* __restrict__ vb, int T) {
  const int bt   = blockIdx.x;       // b*T + t
  const int b    = bt / T;
  const int t    = bt - b * T;
  const int lane = threadIdx.x & 31;
  const int wave = threadIdx.x >> 5;

  const float c = cosw[t * 32 + lane];
  const float s = sinw[t * 32 + lane];

  // 16 query heads: two rounds of 8 waves.
#pragma unroll
  for (int rr = 0; rr < 2; ++rr) {
    const int h = wave + rr * 8;
    const float* src = qf + (size_t)bt * 1024 + h * 64;
    float e1 = src[lane], e2 = src[lane + 32];
    float ss = e1 * e1 + e2 * e2;
#pragma unroll
    for (int off = 1; off <= 16; off <<= 1) ss += __shfl_xor(ss, off, 32);
    const float r = rsqrtf(ss * (1.0f / 64.0f) + 1e-6f);
    const float n1 = e1 * r * qn[lane];
    const float n2 = e2 * r * qn[lane + 32];
    __bf16* dst = qb + ((size_t)(b * 16 + h) * T + t) * 64;
    dst[lane]      = (__bf16)(n1 * c - n2 * s);
    dst[lane + 32] = (__bf16)(n2 * c + n1 * s);
  }

  if (wave < 4) {  // 4 kv heads: K path (norm + rope)
    const int h = wave;
    const float* src = kf + (size_t)bt * 256 + h * 64;
    float e1 = src[lane], e2 = src[lane + 32];
    float ss = e1 * e1 + e2 * e2;
#pragma unroll
    for (int off = 1; off <= 16; off <<= 1) ss += __shfl_xor(ss, off, 32);
    const float r = rsqrtf(ss * (1.0f / 64.0f) + 1e-6f);
    const float n1 = e1 * r * kn[lane];
    const float n2 = e2 * r * kn[lane + 32];
    __bf16* dst = kb + ((size_t)(b * 4 + h) * T + t) * 64;
    dst[lane]      = (__bf16)(n1 * c - n2 * s);
    dst[lane + 32] = (__bf16)(n2 * c + n1 * s);
  } else {         // V path: straight convert/repack
    const int h = wave - 4;
    const float* src = vf + (size_t)bt * 256 + h * 64;
    __bf16* dst = vb + ((size_t)(b * 4 + h) * T + t) * 64;
    dst[lane]      = (__bf16)src[lane];
    dst[lane + 32] = (__bf16)src[lane + 32];
  }
}

// ---------------------------------------------------------------------------
// Causal flash attention, GQA (q head h -> kv head h/4), D=64, scale=1/8.
// Block: 256 threads = 8 waves; 128 q rows per block (16 per wave).
// K tiles of 32 rows staged by the Tensor Data Mover (wave 0 issues one
// tensor_load_to_lds per tile, pad feature produces 72-half padded rows);
// V tiles staged transposed by all threads. S: 2x2 WMMA; PV: 4 WMMA;
// online softmax in f32 in the C-fragment layout.
// ---------------------------------------------------------------------------
__global__ __launch_bounds__(256) void flash_attn_wmma(
    const __bf16* __restrict__ Q, const __bf16* __restrict__ Kh,
    const __bf16* __restrict__ Vh, __bf16* __restrict__ O, int T) {
  constexpr float kScale = 0.125f;   // 64^-0.5
  constexpr float kNegBig = -3.0e38f;

  __shared__ __bf16 sK[32][72];       // K tile, row = kv, col = d (8-half pad)
  __shared__ __bf16 sVt[64][40];      // V tile transposed: row = d, col = kv
  __shared__ __bf16 sP[8][16][40];    // per-wave P tile 16x32 (bf16)

  const int tid  = threadIdx.x;
  const int lane = tid & 31;
  const int wave = tid >> 5;
  const int hgrp = lane >> 4;
  const int l16  = lane & 15;

  const int q0  = blockIdx.x * 128;
  const int bh  = blockIdx.y;          // b*16 + h
  const int b   = bh >> 4;
  const int h   = bh & 15;
  const int kvh = h >> 2;              // GROUPS = 4

  const __bf16* qp = Q  + (size_t)(b * 16 + h)  * T * 64;
  const __bf16* kp = Kh + (size_t)(b * 4 + kvh) * T * 64;
  const __bf16* vp = Vh + (size_t)(b * 4 + kvh) * T * 64;

  // Q A-fragments (16 rows x 64 d  ->  two 16x32 fragments), from global.
  BFrag qa[2];
  {
    const int qr = q0 + wave * 16 + l16;
#pragma unroll
    for (int df = 0; df < 2; ++df) {
      const __bf16* base = qp + (size_t)qr * 64 + df * 32;
      qa[df].q[0] = *(const uint4*)(base + hgrp * 8);
      qa[df].q[1] = *(const uint4*)(base + 16 + hgrp * 8);
    }
  }

  v8f o[4];
#pragma unroll
  for (int dt = 0; dt < 4; ++dt) o[dt] = (v8f){};
  float mrun[8], lrun[8];
#pragma unroll
  for (int v = 0; v < 8; ++v) { mrun[v] = kNegBig; lrun[v] = 0.0f; }

  const int kend = q0 + 128;           // causal: only tiles up to the diagonal
  for (int k0 = 0; k0 < kend; k0 += 32) {
    __syncthreads();                   // previous-iteration LDS reads done
#if USE_TDM
    if (wave == 0) {
      // TDM: 32x64 bf16 K tile; LDS pad 4 DWORDs (code 3) every 32 DWORDs
      // (code 4) -> 72-half padded rows matching sK's layout.
      tdm_load_tile_2d_bf16(
          (unsigned)(uintptr_t)&sK[0][0],
          (unsigned long long)(uintptr_t)(kp + (size_t)k0 * 64),
          /*tensor_d0=*/64u, /*tensor_d1=*/(unsigned)T,
          /*tile_d0=*/64u, /*tile_d1=*/32u, /*stride_d0=*/64u,
          /*pad_interval_code=*/4u, /*pad_amount_code=*/3u, /*pad_enable=*/1u);
      __builtin_amdgcn_s_wait_tensorcnt(0);
    }
#endif
    {
      // Stage V transposed (and K manually if no TDM); 8 halves per thread.
      const int r  = tid >> 3;          // 0..31 kv row
      const int cc = (tid & 7) * 8;     // 0..56 d col
#if !USE_TDM
      *(uint4*)&sK[r][cc] = *(const uint4*)(kp + (size_t)(k0 + r) * 64 + cc);
#endif
      const __bf16* vrow = vp + (size_t)(k0 + r) * 64 + cc;
#pragma unroll
      for (int e = 0; e < 8; ++e) sVt[cc + e][r] = vrow[e];
    }
    __syncthreads();

    // S = Q K^T : two 16x16 column tiles, each reduced over d (2 WMMAs).
    v8f sacc[2];
#pragma unroll
    for (int nt = 0; nt < 2; ++nt) {
      v8f sv = (v8f){};
#pragma unroll
      for (int df = 0; df < 2; ++df) {
        BFrag bk;
        const int r = nt * 16 + l16;                   // kv row in tile (N)
        bk.q[0] = *(const uint4*)&sK[r][df * 32 + hgrp * 8];
        bk.q[1] = *(const uint4*)&sK[r][df * 32 + 16 + hgrp * 8];
        sv = wmma_bf16(qa[df].v, bk.v, sv);
      }
      sacc[nt] = sv;
    }

    // Online softmax in C-fragment layout; rows live in 16-lane half groups.
#pragma unroll
    for (int v = 0; v < 8; ++v) {
      const int qr = q0 + wave * 16 + v + 8 * hgrp;
      float e0 = sacc[0][v] * kScale;
      float e1 = sacc[1][v] * kScale;
      if (k0 + l16 > qr)      e0 = kNegBig;
      if (k0 + 16 + l16 > qr) e1 = kNegBig;
      float mx = fmaxf(e0, e1);
#pragma unroll
      for (int off = 1; off <= 8; off <<= 1) mx = fmaxf(mx, __shfl_xor(mx, off, 32));
      const float mnew  = fmaxf(mrun[v], mx);
      const float alpha = __expf(mrun[v] - mnew);
      const float p0 = __expf(e0 - mnew);
      const float p1 = __expf(e1 - mnew);
      float rs = p0 + p1;
#pragma unroll
      for (int off = 1; off <= 8; off <<= 1) rs += __shfl_xor(rs, off, 32);
      lrun[v] = lrun[v] * alpha + rs;
      mrun[v] = mnew;
#pragma unroll
      for (int dt = 0; dt < 4; ++dt) o[dt][v] = o[dt][v] * alpha;
      sP[wave][v + 8 * hgrp][l16]      = (__bf16)p0;   // per-wave region
      sP[wave][v + 8 * hgrp][16 + l16] = (__bf16)p1;
    }

    // PV: A = P (16x32, from per-wave LDS), B = V^T tile (32 kv x 16 d).
    BFrag pa;
    pa.q[0] = *(const uint4*)&sP[wave][l16][hgrp * 8];
    pa.q[1] = *(const uint4*)&sP[wave][l16][16 + hgrp * 8];
#pragma unroll
    for (int dt = 0; dt < 4; ++dt) {
      BFrag bv;
      const int d = dt * 16 + l16;                     // output col (N = d)
      bv.q[0] = *(const uint4*)&sVt[d][hgrp * 8];      // kv = hgrp*8 .. +8
      bv.q[1] = *(const uint4*)&sVt[d][16 + hgrp * 8];
      o[dt] = wmma_bf16(pa.v, bv.v, o[dt]);
    }
  }

  // Normalize and store to (B, T, H*64) bf16.
#pragma unroll
  for (int dt = 0; dt < 4; ++dt) {
#pragma unroll
    for (int v = 0; v < 8; ++v) {
      const int qr = q0 + wave * 16 + v + 8 * hgrp;
      const float val = o[dt][v] / lrun[v];
      O[((size_t)(b * T + qr)) * 1024 + h * 64 + dt * 16 + l16] = (__bf16)val;
    }
  }
}

// ---------------------------------------------------------------------------
// Host-side orchestration.
// ---------------------------------------------------------------------------
extern "C" void kernel_launch(void* const* d_in, const int* in_sizes, int n_in,
                              void* d_out, int out_size, void* d_ws, size_t ws_size,
                              hipStream_t stream) {
  const float* x    = (const float*)d_in[0];
  const float* cosw = (const float*)d_in[1];
  const float* sinw = (const float*)d_in[2];
  const float* wq   = (const float*)d_in[3];
  const float* wk   = (const float*)d_in[4];
  const float* wv   = (const float*)d_in[5];
  const float* wo   = (const float*)d_in[6];
  const float* qn   = (const float*)d_in[7];
  const float* kn   = (const float*)d_in[8];
  float* out = (float*)d_out;

  const int B = 2, T = 2048, DM = 1024, NKV = 4;
  const int M = B * T;                 // 4096

  char* wsp = (char*)d_ws;
  auto alloc = [&](size_t bytes) -> char* {
    char* p = wsp;
    wsp += (bytes + 255) & ~(size_t)255;
    return p;
  };
  __bf16* xb  = (__bf16*)alloc((size_t)M * DM * 2);
  __bf16* wqb = (__bf16*)alloc((size_t)DM * DM * 2);
  __bf16* wkb = (__bf16*)alloc((size_t)NKV * 64 * DM * 2);
  __bf16* wvb = (__bf16*)alloc((size_t)NKV * 64 * DM * 2);
  __bf16* wob = (__bf16*)alloc((size_t)DM * DM * 2);
  float*  qf  = (float*)alloc((size_t)M * DM * 4);
  float*  kf  = (float*)alloc((size_t)M * 256 * 4);
  float*  vf  = (float*)alloc((size_t)M * 256 * 4);
  __bf16* qb  = (__bf16*)alloc((size_t)M * DM * 2);      // (B,16,T,64)
  __bf16* kb  = (__bf16*)alloc((size_t)M * 256 * 2);     // (B,4,T,64)
  __bf16* vb  = (__bf16*)alloc((size_t)M * 256 * 2);     // (B,4,T,64)
  __bf16* ob  = (__bf16*)alloc((size_t)M * DM * 2);      // attn out (B,T,1024)

  auto cvt = [&](const float* src, __bf16* dst, int n) {
    cvt_f32_bf16<<<(n + 255) / 256, 256, 0, stream>>>(src, dst, n);
  };
  cvt(x,  xb,  M * DM);
  cvt(wq, wqb, DM * DM);
  cvt(wk, wkb, 256 * DM);
  cvt(wv, wvb, 256 * DM);
  cvt(wo, wob, DM * DM);

  // Projections: q (4096x1024), k/v (4096x256), all K=1024.
  gemm_bf16_wmma<<<dim3(DM / 128, M / 128), 256, 0, stream>>>(xb, wqb, qf, M, DM, DM);
  gemm_bf16_wmma<<<dim3(256 / 128, M / 128), 256, 0, stream>>>(xb, wkb, kf, M, 256, DM);
  gemm_bf16_wmma<<<dim3(256 / 128, M / 128), 256, 0, stream>>>(xb, wvb, vf, M, 256, DM);

  // RMSNorm + RoPE + pack to head-major bf16.
  norm_rope_pack<<<M, 256, 0, stream>>>(qf, kf, vf, cosw, sinw, qn, kn, qb, kb, vb, T);

  // Causal GQA flash attention.
  flash_attn_wmma<<<dim3(T / 128, B * 16), 256, 0, stream>>>(qb, kb, vb, ob, T);

  // Output projection -> d_out (f32).
  gemm_bf16_wmma<<<dim3(DM / 128, M / 128), 256, 0, stream>>>(ob, wob, out, M, DM, DM);
}
